// GCRNN_5059471474819
// MI455X (gfx1250) — matrix operand order repaired and checked by reading, not compiled, not run on hardware
//
#include <hip/hip_runtime.h>
#include <hip/hip_bf16.h>
#include <stdint.h>

// ---------------------------------------------------------------------------
// GCRNN for MI455X (gfx1250, wave32).
// ~97% of the ~60 GFLOP is dense 2048x2048 graph-mix GEMMs -> run those on
// v_wmma_f32_16x16x32_bf16 (bf16 in, f32 accumulate).  The 8 MB bf16
// weighting matrix is L2-resident (192 MB L2); per K-chunk its 8 KB A-tile is
// DMA'd into LDS by the Tensor Data Mover (tensor_load_to_lds, TENSORcnt),
// double-buffered so TDM of chunk k+1 overlaps the WMMA of chunk k.
// Each of the 8 waves owns a 32x32 output tile (4 accumulators) of the
// 128x64 block tile.  Activations are fused into the WMMA epilogues.
// Small channel matmuls (K<=128) stay exact f32 on VALU.
// ---------------------------------------------------------------------------

typedef __attribute__((ext_vector_type(16))) __bf16 v16bf;
typedef __attribute__((ext_vector_type(8)))  float  v8f;

#define N_NODES 2048
#define D_IN    32
#define H_DIM   64
#define T_STEPS 8
#define B_BATCH 4
#define NCLASS  8

#ifndef __has_builtin
#define __has_builtin(x) 0
#endif
#if defined(__AMDGCN__) && __has_builtin(__builtin_amdgcn_tensor_load_to_lds) && \
    __has_builtin(__builtin_amdgcn_s_wait_tensorcnt)
#define HAVE_TDM 1
#else
#define HAVE_TDM 0
#endif

__device__ __forceinline__ unsigned short f2bf(float f) {
  union { float f; uint32_t u; } v; v.f = f;
  uint32_t u = v.u;
  uint32_t r = u + 0x7FFFu + ((u >> 16) & 1u);   // round-to-nearest-even
  return (unsigned short)(r >> 16);
}

// --------------------------- tiny utility kernels ---------------------------

__global__ void prep_weight_kernel(const float* __restrict__ S,
                                   const float* __restrict__ adj,
                                   unsigned short* __restrict__ Wgt, int n) {
  int i = blockIdx.x * blockDim.x + threadIdx.x;
  if (i < n) Wgt[i] = f2bf(S[i] * adj[i]);
}

__global__ void zero_kernel(float* __restrict__ p, int n) {
  int i = blockIdx.x * blockDim.x + threadIdx.x;
  if (i < n) p[i] = 0.0f;
}

// --------------------------- small channel GEMM ------------------------------
// out[b,m,co] = sum_k in[b,m,k] * W[k,co], in built per `mode`:
//   mode 0: concat(x_t, h)   mode 1: concat(x_t, r*h)   mode 2: f32 buffer X
__global__ void small_gemm_kernel(const float* __restrict__ X,
                                  const float* __restrict__ Hb,
                                  const float* __restrict__ Gb,
                                  const float* __restrict__ W,
                                  unsigned short* __restrict__ Out,
                                  int Cin, int Cout, int mode, int t) {
  int b   = blockIdx.y;
  int gid = blockIdx.x * blockDim.x + threadIdx.x;
  int m   = gid / Cout;
  int co  = gid % Cout;
  size_t bm = (size_t)b * N_NODES + m;
  const float* xt = X + (((size_t)b * T_STEPS + t) * N_NODES + m) * D_IN;

  float acc = 0.0f;
  for (int k = 0; k < Cin; ++k) {
    float in;
    if (mode == 2) {
      in = X[bm * Cin + k];
    } else if (k < D_IN) {
      in = xt[k];
    } else {
      float hv = Hb[bm * H_DIM + (k - D_IN)];
      in = (mode == 0) ? hv : Gb[bm * 128 + (k - D_IN)] * hv;  // r gate
    }
    acc += in * W[(size_t)k * Cout + co];
  }
  Out[bm * Cout + co] = f2bf(acc);
}

// --------------------------- big WMMA graph GEMM -----------------------------
// Out[b] = act( Wgt(2048x2048,bf16) @ Bin[b](2048xC,bf16) ), C compile-time.
// ACT: 0 identity | 1 sigmoid | 2 tanh + GRU update (in-place h) | 3 leaky.

union FragAB { v16bf v; uint4 q[2]; };
union FragC  { v8f   v; float f[8]; };

__device__ __forceinline__ v8f wmma_bf16(const FragAB& a, const FragAB& b,
                                         const FragC& c) {
  return __builtin_amdgcn_wmma_f32_16x16x32_bf16(
      false, a.v, false, b.v, (short)0, c.v, false, false);
}

#if HAVE_TDM
typedef __attribute__((ext_vector_type(4))) unsigned int v4u_t;
typedef __attribute__((ext_vector_type(8))) int          v8i_t;
typedef __attribute__((ext_vector_type(4))) int          v4i_t;

// Issue one TDM descriptor: load a 128x32-element bf16 tile (rows stride 2048)
// from Wgt into LDS at `ldsDst` (rows packed contiguously, 64 B each).
// D# bitfields per CDNA5 ISA 8.3/8.4 (2-D tensor; groups 2/3 zero).
__device__ __forceinline__ void tdm_issue_a(const unsigned short* __restrict__ Wgt,
                                            int mblk, int kc, void* ldsDst) {
  uint64_t gaddr = (uint64_t)(uintptr_t)(Wgt + (size_t)mblk * N_NODES + kc);
  uint32_t laddr = (uint32_t)(uintptr_t)ldsDst;   // addr[31:0] == LDS offset
  v4u_t g0;
  g0[0] = 1u;                                          // count=1, user mode
  g0[1] = laddr;                                       // lds_addr [63:32]
  g0[2] = (uint32_t)gaddr;                             // global_addr [95:64]
  g0[3] = (uint32_t)((gaddr >> 32) & 0x01FFFFFFu)      // global_addr [120:96]
        | (2u << 30);                                  // type=2 [127:126]
  v8i_t g1;
  g1[0] = (int)(1u << 16);            // workgroup_mask=0, data_size=1 (2 B)
  g1[1] = (int)((uint32_t)N_NODES << 16);  // tensor_dim0[15:0] @ [63:48]
  g1[2] = (int)((uint32_t)N_NODES << 16);  // tensor_dim1[15:0] @ [95:80]
  g1[3] = (int)(32u << 16);           // tile_dim0=32 @ [127:112]
  g1[4] = 128;                        // tile_dim1=128 @ [143:128]
  g1[5] = N_NODES;                    // tensor_dim0_stride[31:0]
  g1[6] = 0;
  g1[7] = 0;
  v4i_t z4 = {0, 0, 0, 0};
  v8i_t z8 = {0, 0, 0, 0, 0, 0, 0, 0};
  // 6-arg form (clang-23 / therock-10.0 headers): groups 2/3 + trailing group
  // are unused for a 2-D tensor.
  __builtin_amdgcn_tensor_load_to_lds(g0, g1, z4, z4, z8, 0);
}
#endif

// Fallback / host-pass A staging: 128x32 bf16 via 16 B vector copies.
__device__ __forceinline__ void stage_a(unsigned short (*as)[32],
                                        const unsigned short* __restrict__ Wgt,
                                        int mblk, int kc, int tid) {
#pragma unroll
  for (int i = 0; i < 2; ++i) {
    int idx = tid + i * 256;            // 0..511
    int row = idx >> 2, q = idx & 3;
    *(uint4*)&as[row][q * 8] =
        *(const uint4*)(Wgt + (size_t)(mblk + row) * N_NODES + kc + q * 8);
  }
}

// Stage B transposed: bt[n][k] = Bin[kc+k][nb+n], zero-padded past C.
template <int C>
__device__ __forceinline__ void stage_b(unsigned short (*bt)[32],
                                        const unsigned short* __restrict__ Bb,
                                        int kc, int nb, int tid) {
#pragma unroll
  for (int i = 0; i < 8; ++i) {
    int idx = tid + i * 256;            // 0..2047
    int k = idx >> 6;                   // 0..31
    int n = idx & 63;                   // 0..63
    unsigned short v = 0;
    if (C >= 64 || n < C) v = Bb[(size_t)(kc + k) * C + nb + n];
    bt[n][k] = v;
  }
}

template <int ACT, int C>
__global__ __launch_bounds__(256)
void big_gemm_kernel(const unsigned short* __restrict__ Wgt,
                     const unsigned short* __restrict__ Bin,
                     float* __restrict__ Out,
                     const float* __restrict__ Gbuf) {
  __shared__ __align__(16) unsigned short As[2][128][32];  // 16 KB, dbl-buffered
  __shared__ __align__(16) unsigned short Bt[2][64][32];   //  8 KB, [n][k]

  const int tid  = threadIdx.x;
  const int b    = blockIdx.z;
  const int mblk = blockIdx.x * 128;
  const int nb   = blockIdx.y * 64;

  const unsigned short* Bb = Bin + (size_t)b * N_NODES * C;
  float*                Ob = Out + (size_t)b * N_NODES * C;
  const float*          Gb = Gbuf + (size_t)b * N_NODES * 128;  // ACT==2 only

  const int w    = tid >> 5;          // wave 0..7 (wave32)
  const int lane = tid & 31;
  const int l15  = lane & 15;
  const int hi   = lane >> 4;
  const int m0   = (w >> 1) * 32;     // wave's 32-row sub-tile
  const int n0   = (w & 1) * 32;      // wave's 32-col sub-tile

  FragC acc[2][2];
#pragma unroll
  for (int mi = 0; mi < 2; ++mi)
#pragma unroll
    for (int ni = 0; ni < 2; ++ni)
#pragma unroll
      for (int r = 0; r < 8; ++r) acc[mi][ni].f[r] = 0.0f;

  constexpr int NCH = N_NODES / 32;   // 64 K-chunks

  // Prologue: kick off chunk 0 into buffer 0.
#if HAVE_TDM
  if (w == 0) tdm_issue_a(Wgt, mblk, 0, &As[0][0][0]);
#else
  stage_a(As[0], Wgt, mblk, 0, tid);
#endif
  stage_b<C>(Bt[0], Bb, 0, nb, tid);

  for (int i = 0; i < NCH; ++i) {
    const int buf = i & 1;
#if HAVE_TDM
    if (w == 0) __builtin_amdgcn_s_wait_tensorcnt(0);  // A(buf) landed in LDS
#endif
    __syncthreads();                                    // A(buf),B(buf) visible
    if (i + 1 < NCH) {                                  // prefetch next chunk
#if HAVE_TDM
      if (w == 0) tdm_issue_a(Wgt, mblk, (i + 1) * 32, &As[buf ^ 1][0][0]);
#else
      stage_a(As[buf ^ 1], Wgt, mblk, (i + 1) * 32, tid);
#endif
      stage_b<C>(Bt[buf ^ 1], Bb, (i + 1) * 32, nb, tid);
    }

    const unsigned short(*A)[32] = As[buf];
    const unsigned short(*Bw)[32] = Bt[buf];
    // 16-bit WMMA fragment layouts: two ds_load_b128 per fragment.
    FragAB a0, a1, b0, b1;
    a0.q[0] = *(const uint4*)&A[m0 + l15][8 * hi];
    a0.q[1] = *(const uint4*)&A[m0 + l15][16 + 8 * hi];
    a1.q[0] = *(const uint4*)&A[m0 + 16 + l15][8 * hi];
    a1.q[1] = *(const uint4*)&A[m0 + 16 + l15][16 + 8 * hi];
    b0.q[0] = *(const uint4*)&Bw[n0 + l15][8 * hi];
    b0.q[1] = *(const uint4*)&Bw[n0 + l15][16 + 8 * hi];
    b1.q[0] = *(const uint4*)&Bw[n0 + 16 + l15][8 * hi];
    b1.q[1] = *(const uint4*)&Bw[n0 + 16 + l15][16 + 8 * hi];

    acc[0][0].v = wmma_bf16(a0, b0, acc[0][0]);
    acc[0][1].v = wmma_bf16(a0, b1, acc[0][1]);
    acc[1][0].v = wmma_bf16(a1, b0, acc[1][0]);
    acc[1][1].v = wmma_bf16(a1, b1, acc[1][1]);
    __syncthreads();
  }

  // Epilogue: element r of C/D frag -> row = r + 8*hi, col = lane&15.
#pragma unroll
  for (int ni = 0; ni < 2; ++ni) {
    const int coln = nb + n0 + ni * 16 + l15;
    if (C >= 64 || coln < C) {
#pragma unroll
      for (int mi = 0; mi < 2; ++mi) {
#pragma unroll
        for (int r = 0; r < 8; ++r) {
          int row = mblk + m0 + mi * 16 + r + 8 * hi;
          float v = acc[mi][ni].f[r];
          if (ACT == 1) {
            v = 1.0f / (1.0f + expf(-v));
          } else if (ACT == 3) {
            v = (v >= 0.0f) ? v : 0.01f * v;
          } else if (ACT == 2) {
            float u  = Gb[(size_t)row * 128 + 64 + coln];  // update gate
            float ho = Ob[(size_t)row * C + coln];         // old h (C==64)
            v = u * ho + (1.0f - u) * tanhf(v);
          }
          Ob[(size_t)row * C + coln] = v;
        }
      }
    }
  }
}

// ------------------------------- launcher ------------------------------------

extern "C" void kernel_launch(void* const* d_in, const int* in_sizes, int n_in,
                              void* d_out, int out_size, void* d_ws, size_t ws_size,
                              hipStream_t stream) {
  (void)in_sizes; (void)n_in; (void)out_size; (void)ws_size;

  const float* inputs = (const float*)d_in[0];  // (B,T,N,D)
  const float* adj    = (const float*)d_in[1];  // (N,N)
  const float* S      = (const float*)d_in[2];  // (N,N)
  const float* W_gate = (const float*)d_in[3];  // (96,128)
  const float* W_cand = (const float*)d_in[4];  // (96,64)
  const float* W_gc1  = (const float*)d_in[5];  // (64,128)
  const float* W_gc2  = (const float*)d_in[6];  // (128,64)
  const float* W_last = (const float*)d_in[7];  // (64,8)
  float* out = (float*)d_out;                   // (B,N,8)

  // Workspace partition (~21 MB total)
  char* ws = (char*)d_ws;
  unsigned short* Wgt   = (unsigned short*)(ws);                // 8 MB bf16 NxN
  float*          hbuf  = (float*)(ws + (8u  << 20));           // 2 MB  B*N*64
  float*          gbuf  = (float*)(ws + (10u << 20));           // 4 MB  B*N*128
  unsigned short* pbuf  = (unsigned short*)(ws + (14u << 20));  // 2 MB  B*N*128 bf16
  unsigned short* h2buf = (unsigned short*)(ws + (16u << 20));  // 1 MB  B*N*64 bf16
  float*          xbuf  = (float*)(ws + (17u << 20));           // 4 MB  B*N*128

  dim3 blk(256);

  prep_weight_kernel<<<(N_NODES * N_NODES) / 256, blk, 0, stream>>>(
      S, adj, Wgt, N_NODES * N_NODES);
  zero_kernel<<<(B_BATCH * N_NODES * H_DIM) / 256, blk, 0, stream>>>(
      hbuf, B_BATCH * N_NODES * H_DIM);

  // ------------------------------ GRU scan ------------------------------
  for (int t = 0; t < T_STEPS; ++t) {
    small_gemm_kernel<<<dim3((N_NODES * 128) / 256, B_BATCH), blk, 0, stream>>>(
        inputs, hbuf, nullptr, W_gate, pbuf, D_IN + H_DIM, 128, 0, t);
    big_gemm_kernel<1, 128><<<dim3(16, 2, B_BATCH), blk, 0, stream>>>(
        Wgt, pbuf, gbuf, gbuf);
    small_gemm_kernel<<<dim3((N_NODES * H_DIM) / 256, B_BATCH), blk, 0, stream>>>(
        inputs, hbuf, gbuf, W_cand, h2buf, D_IN + H_DIM, H_DIM, 1, t);
    big_gemm_kernel<2, 64><<<dim3(16, 1, B_BATCH), blk, 0, stream>>>(
        Wgt, h2buf, hbuf, gbuf);
  }

  // ------------------------------ GC head -------------------------------
  small_gemm_kernel<<<dim3((N_NODES * 128) / 256, B_BATCH), blk, 0, stream>>>(
      hbuf, nullptr, nullptr, W_gc1, pbuf, H_DIM, 128, 2, 0);
  big_gemm_kernel<3, 128><<<dim3(16, 2, B_BATCH), blk, 0, stream>>>(
      Wgt, pbuf, xbuf, gbuf);
  small_gemm_kernel<<<dim3((N_NODES * H_DIM) / 256, B_BATCH), blk, 0, stream>>>(
      xbuf, nullptr, nullptr, W_gc2, h2buf, 128, H_DIM, 2, 0);
  big_gemm_kernel<3, 64><<<dim3(16, 1, B_BATCH), blk, 0, stream>>>(
      Wgt, h2buf, xbuf, gbuf);
  small_gemm_kernel<<<dim3((N_NODES * NCLASS) / 256, B_BATCH), blk, 0, stream>>>(
      xbuf, nullptr, nullptr, W_last, pbuf, H_DIM, NCLASS, 2, 0);
  big_gemm_kernel<0, NCLASS><<<dim3(16, 1, B_BATCH), blk, 0, stream>>>(
      Wgt, pbuf, out, gbuf);
}
